// MPModel_45964740002216
// MI455X (gfx1250) — matrix operand order repaired and checked by reading, not compiled, not run on hardware
//
#include <hip/hip_runtime.h>
#include <cstddef>
#include <cstdint>

#define DEV __device__ __forceinline__

typedef __attribute__((ext_vector_type(16))) __bf16 bf16x16;
typedef __attribute__((ext_vector_type(8)))  __bf16 bf16x8;
typedef __attribute__((ext_vector_type(8)))  float  f32x8;

// ---------------------------------------------------------------------------
// WMMA helpers (CDNA5 v_wmma_f32_16x16x32_bf16, wave32)
// ---------------------------------------------------------------------------

DEV bf16x16 join8(bf16x8 lo, bf16x8 hi) {
  return __builtin_shufflevector(lo, hi, 0,1,2,3,4,5,6,7,8,9,10,11,12,13,14,15);
}

// A fragment: 16x32 bf16 tile from row-major bf16 matrix (leading dim ldk).
// Lanes 0-15: row = lane, K = {kc..kc+7, kc+16..kc+23}
// Lanes 16-31: row = lane-16, K = {kc+8..kc+15, kc+24..kc+31}
DEV bf16x16 frag_a_bf16(const __bf16* A, int ldk, int row0, int kc) {
  int lane = threadIdx.x & 31;
  int r = lane & 15, h = lane >> 4;
  const __bf16* p = A + (size_t)(row0 + r) * ldk + kc + h * 8;
  bf16x8 lo = *(const bf16x8*)p;
  bf16x8 hi = *(const bf16x8*)(p + 16);
  return join8(lo, hi);
}

// A fragment from fp32 source, converted inline to bf16.
DEV bf16x16 frag_a_f32(const float* A, int ldk, int row0, int kc) {
  int lane = threadIdx.x & 31;
  int r = lane & 15, h = lane >> 4;
  const float* p = A + (size_t)(row0 + r) * ldk + kc + h * 8;
  bf16x16 v;
#pragma unroll
  for (int j = 0; j < 8; ++j) v[j] = (__bf16)p[j];
#pragma unroll
  for (int j = 0; j < 8; ++j) v[8 + j] = (__bf16)p[16 + j];
  return v;
}

// B fragment: 32x16 tile where B[k][n] = W[n][k], W row-major bf16 [n, ldk].
// Lanes 0-15: col n0+lane, K = {kc..kc+7, kc+16..kc+23}; lanes 16-31: K = {kc+8..kc+15, kc+24..kc+31}.
DEV bf16x16 frag_b_w(const __bf16* W, int ldk, int n0, int kc) {
  int lane = threadIdx.x & 31;
  int c = lane & 15, h = lane >> 4;
  const __bf16* p = W + (size_t)(n0 + c) * ldk + kc + h * 8;
  return join8(*(const bf16x8*)p, *(const bf16x8*)(p + 16));
}

DEV f32x8 wmma_bf16(bf16x16 a, bf16x16 b, f32x8 c) {
  return __builtin_amdgcn_wmma_f32_16x16x32_bf16(
      /*neg_a=*/false, a, /*neg_b=*/false, b,
      /*c_mod=*/(short)0, c, /*reuse_a=*/false, /*reuse_b=*/false);
}

DEV float bf2f(unsigned short s) {
  unsigned u = ((unsigned)s) << 16;
  float f;
  __builtin_memcpy(&f, &u, 4);
  return f;
}

// ---------------------------------------------------------------------------
// Prep kernels
// ---------------------------------------------------------------------------

__global__ void k_f32_to_bf16(const float* __restrict__ in, __bf16* __restrict__ out, int n) {
  int i = blockIdx.x * blockDim.x + threadIdx.x;
  if (i < n) out[i] = (__bf16)in[i];
}

// Split msg_w [64 x 128] into wA (first 64 cols) / wB (last 64), transpose root.
__global__ void k_prep_small(const float* __restrict__ msgw, const float* __restrict__ root,
                             __bf16* __restrict__ wA, __bf16* __restrict__ wB,
                             __bf16* __restrict__ rT) {
  int i = blockIdx.x * blockDim.x + threadIdx.x;
  if (i >= 64 * 64) return;
  int o = i >> 6, k = i & 63;
  wA[i] = (__bf16)msgw[o * 128 + k];
  wB[i] = (__bf16)msgw[o * 128 + 64 + k];
  rT[i] = (__bf16)root[k * 64 + o];  // rT[o,k] = root[k,o]
}

__global__ void k_zero(float* __restrict__ p, int n) {
  int i = blockIdx.x * blockDim.x + threadIdx.x;
  if (i < n) p[i] = 0.0f;
}

// ---------------------------------------------------------------------------
// out0 = relu(nf @ mp_w^T + mp_b)  -> bf16   [N x 64]
// ---------------------------------------------------------------------------
__global__ void k_node_in(const float* __restrict__ nf, const __bf16* __restrict__ mpw,
                          const float* __restrict__ mpb, __bf16* __restrict__ out_b,
                          int mtiles) {
  int wave = (blockIdx.x * blockDim.x + threadIdx.x) >> 5;
  if (wave >= mtiles) return;  // wave-uniform: EXEC all-ones for WMMA
  int row0 = wave * 16;
  int lane = threadIdx.x & 31, c = lane & 15, rb = (lane >> 4) * 8;
  f32x8 acc[4] = {};
  for (int kc = 0; kc < 64; kc += 32) {
    bf16x16 a = frag_a_f32(nf, 64, row0, kc);
#pragma unroll
    for (int t = 0; t < 4; ++t)
      acc[t] = wmma_bf16(a, frag_b_w(mpw, 64, t * 16, kc), acc[t]);
  }
#pragma unroll
  for (int t = 0; t < 4; ++t) {
    int col = t * 16 + c;
    float bias = mpb[col];
#pragma unroll
    for (int j = 0; j < 8; ++j) {
      float v = acc[t][j] + bias;
      v = v > 0.0f ? v : 0.0f;
      out_b[(size_t)(row0 + rb + j) * 64 + col] = (__bf16)v;
    }
  }
}

// ---------------------------------------------------------------------------
// h1 = relu(edge_attr @ e1_w^T + e1_b) -> bf16   [E x 128]  (K=16: VALU)
// ---------------------------------------------------------------------------
__global__ void k_edge_h1(const float* __restrict__ ea, const float* __restrict__ e1w,
                          const float* __restrict__ e1b, __bf16* __restrict__ h1, int E) {
  int e = blockIdx.x * 2 + (threadIdx.x >> 7);
  int j = threadIdx.x & 127;
  if (e >= E) return;
  const float* a = ea + (size_t)e * 16;
  const float* w = e1w + (size_t)j * 16;
  float acc = e1b[j];
#pragma unroll
  for (int k = 0; k < 16; ++k) acc += a[k] * w[k];
  acc = acc > 0.0f ? acc : 0.0f;
  h1[(size_t)e * 128 + j] = (__bf16)acc;
}

// ---------------------------------------------------------------------------
// W_e = h1 @ e2_w^T + e2_b -> bf16   [E x 4096]   (the 105-GFLOP GEMM)
// One wave computes a 16(M) x 64(N) tile, K = 128 in 4 chunks of 32.
// ---------------------------------------------------------------------------
__global__ void k_edge_We(const __bf16* __restrict__ h1, const __bf16* __restrict__ e2w,
                          const float* __restrict__ e2b, __bf16* __restrict__ We,
                          int mtiles) {
  int wave = (blockIdx.x * blockDim.x + threadIdx.x) >> 5;
  int mt = wave >> 6;          // 64 N-blocks of 64 columns
  int nb = wave & 63;
  if (mt >= mtiles) return;
  int row0 = mt * 16, n0 = nb * 64;
  int lane = threadIdx.x & 31, c = lane & 15, rb = (lane >> 4) * 8;
  f32x8 acc[4] = {};
#pragma unroll
  for (int kc = 0; kc < 128; kc += 32) {
    bf16x16 a = frag_a_bf16(h1, 128, row0, kc);
#pragma unroll
    for (int t = 0; t < 4; ++t)
      acc[t] = wmma_bf16(a, frag_b_w(e2w, 128, n0 + t * 16, kc), acc[t]);
  }
#pragma unroll
  for (int t = 0; t < 4; ++t) {
    int col = n0 + t * 16 + c;
    float bias = e2b[col];
#pragma unroll
    for (int j = 0; j < 8; ++j)
      We[(size_t)(row0 + rb + j) * 4096 + col] = (__bf16)(acc[t][j] + bias);
  }
}

// ---------------------------------------------------------------------------
// Per-step: msgs[e] = x[src[e]] @ W_e[e]; agg[dst[e]] += msgs  (one wave/edge)
//
// The 8KB/edge W_e stream is the roofline bottleneck; double-buffer it
// through LDS with gfx1250 async global->LDS DMA (ASYNCcnt-tracked):
//   chunk = 16 rows x 128B = 2KB = 4x global_load_async_to_lds_b128 per wave.
// While chunk c is consumed from LDS, chunk c+1 is in flight; async loads
// complete in order, so s_wait_asynccnt 4 retires exactly the older chunk.
// ---------------------------------------------------------------------------
__global__ void __launch_bounds__(256)
k_msg_scatter(const __bf16* __restrict__ x_b, const __bf16* __restrict__ We,
              const int* __restrict__ src, const int* __restrict__ dst,
              float* __restrict__ agg, int E) {
  __shared__ unsigned smem[8][2][512];  // 8 waves x double buffer x 2KB
  int e = (blockIdx.x * blockDim.x + threadIdx.x) >> 5;
  if (e >= E) return;  // wave-uniform
  int wid = (threadIdx.x >> 5) & 7;
  int lane = threadIdx.x & 31;
  int s = src[e], d = dst[e];

  // gather source row: lane holds x[s, 2*lane], x[s, 2*lane+1]
  unsigned ux = *(const unsigned*)(x_b + (size_t)s * 64 + 2 * lane);
  float xa = bf2f((unsigned short)(ux & 0xffff));
  float xb = bf2f((unsigned short)(ux >> 16));

  unsigned long long gbase =
      (unsigned long long)(uintptr_t)(We + (size_t)e * 4096) + (unsigned)(lane * 16);
  // low 32 bits of a generic LDS pointer == LDS byte offset on gfx9+/gfx12
  unsigned lbase = (unsigned)(uintptr_t)(&smem[wid][0][0]) + (unsigned)(lane * 16);

  // prime chunk 0
#pragma unroll
  for (int q = 0; q < 4; ++q) {
    unsigned long long ga = gbase + q * 512;
    unsigned la = lbase + q * 512;
    asm volatile("global_load_async_to_lds_b128 %0, %1, off"
                 :: "v"(la), "v"(ga) : "memory");
  }

  float acc0 = 0.0f, acc1 = 0.0f;
#pragma unroll
  for (int ch = 0; ch < 4; ++ch) {
    if (ch < 3) {
#pragma unroll
      for (int q = 0; q < 4; ++q) {
        unsigned long long ga = gbase + (unsigned)((ch + 1) * 2048 + q * 512);
        unsigned la = lbase + (unsigned)(((ch + 1) & 1) * 2048 + q * 512);
        asm volatile("global_load_async_to_lds_b128 %0, %1, off"
                     :: "v"(la), "v"(ga) : "memory");
      }
      asm volatile("s_wait_asynccnt 0x4" ::: "memory");  // chunk ch landed
    } else {
      asm volatile("s_wait_asynccnt 0x0" ::: "memory");  // last chunk landed
    }
    const unsigned* buf = &smem[wid][ch & 1][0];
#pragma unroll
    for (int il = 0; il < 16; ++il) {
      int i = ch * 16 + il;
      float xi = __shfl((i & 1) ? xb : xa, i >> 1, 32);  // broadcast x[s,i]
      unsigned uw = buf[il * 32 + lane];                 // ds_load_b32
      acc0 += xi * bf2f((unsigned short)(uw & 0xffff));
      acc1 += xi * bf2f((unsigned short)(uw >> 16));
    }
  }
  float* ap = agg + (size_t)d * 64 + 2 * lane;
  atomicAdd(ap, acc0);
  atomicAdd(ap + 1, acc1);
}

// ---------------------------------------------------------------------------
// m = relu(agg + out @ root + conv_b) -> bf16
// ---------------------------------------------------------------------------
__global__ void k_node_m(const __bf16* __restrict__ out_b, const __bf16* __restrict__ rootT,
                         const float* __restrict__ agg, const float* __restrict__ convb,
                         __bf16* __restrict__ m_b, int mtiles) {
  int wave = (blockIdx.x * blockDim.x + threadIdx.x) >> 5;
  if (wave >= mtiles) return;
  int row0 = wave * 16;
  int lane = threadIdx.x & 31, c = lane & 15, rb = (lane >> 4) * 8;
  f32x8 acc[4] = {};
  for (int kc = 0; kc < 64; kc += 32) {
    bf16x16 a = frag_a_bf16(out_b, 64, row0, kc);
#pragma unroll
    for (int t = 0; t < 4; ++t)
      acc[t] = wmma_bf16(a, frag_b_w(rootT, 64, t * 16, kc), acc[t]);
  }
#pragma unroll
  for (int t = 0; t < 4; ++t) {
    int col = t * 16 + c;
    float bias = convb[col];
#pragma unroll
    for (int j = 0; j < 8; ++j) {
      size_t idx = (size_t)(row0 + rb + j) * 64 + col;
      float v = acc[t][j] + agg[idx] + bias;
      v = v > 0.0f ? v : 0.0f;
      m_b[idx] = (__bf16)v;
    }
  }
}

// ---------------------------------------------------------------------------
// out_new = concat(m, out) @ msg_w^T + msg_b  (= m@wA^T + out@wB^T + b) [+ nf]
// Writes fp32 (d_out on last step) and bf16 (in-place over out_b is safe:
// each wave reads exactly the rows it later writes).
// ---------------------------------------------------------------------------
__global__ void k_node_out(const __bf16* __restrict__ m_b, const __bf16* __restrict__ out_b_in,
                           const __bf16* __restrict__ wA, const __bf16* __restrict__ wB,
                           const float* __restrict__ msgb, const float* __restrict__ add_nf,
                           float* __restrict__ out_f, __bf16* __restrict__ out_b_next,
                           int mtiles) {
  int wave = (blockIdx.x * blockDim.x + threadIdx.x) >> 5;
  if (wave >= mtiles) return;
  int row0 = wave * 16;
  int lane = threadIdx.x & 31, c = lane & 15, rb = (lane >> 4) * 8;
  f32x8 acc[4] = {};
  for (int kc = 0; kc < 64; kc += 32) {
    bf16x16 am = frag_a_bf16(m_b, 64, row0, kc);
    bf16x16 ao = frag_a_bf16(out_b_in, 64, row0, kc);
#pragma unroll
    for (int t = 0; t < 4; ++t) {
      acc[t] = wmma_bf16(am, frag_b_w(wA, 64, t * 16, kc), acc[t]);
      acc[t] = wmma_bf16(ao, frag_b_w(wB, 64, t * 16, kc), acc[t]);
    }
  }
#pragma unroll
  for (int t = 0; t < 4; ++t) {
    int col = t * 16 + c;
    float bias = msgb[col];
#pragma unroll
    for (int j = 0; j < 8; ++j) {
      size_t idx = (size_t)(row0 + rb + j) * 64 + col;
      float v = acc[t][j] + bias;
      if (add_nf) v += add_nf[idx];
      out_f[idx] = v;
      out_b_next[idx] = (__bf16)v;
    }
  }
}

// ---------------------------------------------------------------------------
// Host launcher
// ---------------------------------------------------------------------------
extern "C" void kernel_launch(void* const* d_in, const int* in_sizes, int n_in,
                              void* d_out, int out_size, void* d_ws, size_t ws_size,
                              hipStream_t stream) {
  const float* nf    = (const float*)d_in[0];
  const float* ea    = (const float*)d_in[1];
  const int*   eidx  = (const int*)d_in[2];
  const float* mp_w  = (const float*)d_in[3];
  const float* mp_b  = (const float*)d_in[4];
  const float* msg_w = (const float*)d_in[5];
  const float* msg_b = (const float*)d_in[6];
  const float* e1_w  = (const float*)d_in[7];
  const float* e1_b  = (const float*)d_in[8];
  const float* e2_w  = (const float*)d_in[9];
  const float* e2_b  = (const float*)d_in[10];
  const float* root  = (const float*)d_in[11];
  const float* convb = (const float*)d_in[12];

  const int N = in_sizes[0] / 64;
  const int E = in_sizes[2] / 2;
  const int n_e2w = in_sizes[9];           // 4096*128
  const int* src = eidx;
  const int* dst = eidx + E;

  // workspace carve-up (256B aligned)
  uint8_t* ws = (uint8_t*)d_ws;
  size_t off = 0;
  auto carve = [&](size_t bytes) {
    uint8_t* p = ws + off;
    off = (off + bytes + 255) & ~(size_t)255;
    return p;
  };
  __bf16* We_bf   = (__bf16*)carve((size_t)E * 4096 * 2);
  __bf16* h1_bf   = (__bf16*)carve((size_t)E * 128 * 2);
  __bf16* e2w_bf  = (__bf16*)carve((size_t)n_e2w * 2);
  __bf16* mpw_bf  = (__bf16*)carve(64 * 64 * 2);
  __bf16* wA_bf   = (__bf16*)carve(64 * 64 * 2);
  __bf16* wB_bf   = (__bf16*)carve(64 * 64 * 2);
  __bf16* rT_bf   = (__bf16*)carve(64 * 64 * 2);
  __bf16* out_bf  = (__bf16*)carve((size_t)N * 64 * 2);
  __bf16* m_bf    = (__bf16*)carve((size_t)N * 64 * 2);
  float*  agg     = (float*)carve((size_t)N * 64 * 4);
  float*  outF    = (float*)carve((size_t)N * 64 * 4);
  (void)ws_size; (void)n_in; (void)out_size;

  const int mtilesN = (N + 15) / 16;   // 1250
  const int mtilesE = (E + 15) / 16;   // 6250
  const int nodeBlocks = (mtilesN + 7) / 8;            // 8 waves/block
  const int WeBlocks   = (mtilesE * 64 + 7) / 8;       // 400000 wave-tiles
  const int msgBlocks  = (E + 7) / 8;                  // 8 edges/block
  const int nAgg = N * 64;

  // weight prep
  k_f32_to_bf16<<<(64 * 64 + 255) / 256, 256, 0, stream>>>(mp_w, mpw_bf, 64 * 64);
  k_f32_to_bf16<<<(n_e2w + 255) / 256, 256, 0, stream>>>(e2_w, e2w_bf, n_e2w);
  k_prep_small<<<(64 * 64 + 255) / 256, 256, 0, stream>>>(msg_w, root, wA_bf, wB_bf, rT_bf);

  // out0 = relu(nf @ mp_w^T + mp_b)
  k_node_in<<<nodeBlocks, 256, 0, stream>>>(nf, mpw_bf, mp_b, out_bf, mtilesN);

  // edge-conditioned weights (once; edge_attr constant over steps)
  k_edge_h1<<<(E + 1) / 2, 256, 0, stream>>>(ea, e1_w, e1_b, h1_bf, E);
  k_edge_We<<<WeBlocks, 256, 0, stream>>>(h1_bf, e2w_bf, e2_b, We_bf, mtilesE);

  for (int step = 0; step < 3; ++step) {
    k_zero<<<(nAgg + 255) / 256, 256, 0, stream>>>(agg, nAgg);
    k_msg_scatter<<<msgBlocks, 256, 0, stream>>>(out_bf, We_bf, src, dst, agg, E);
    k_node_m<<<nodeBlocks, 256, 0, stream>>>(out_bf, rT_bf, agg, convb, m_bf, mtilesN);
    const bool last = (step == 2);
    k_node_out<<<nodeBlocks, 256, 0, stream>>>(
        m_bf, out_bf, wA_bf, wB_bf, msg_b,
        last ? nf : nullptr,
        last ? (float*)d_out : outF,
        out_bf, mtilesN);
  }
}
